// GAT_4982162063669
// MI455X (gfx1250) — compile-verified, matrix operand order checked by reading, not run on hardware
//
#include <hip/hip_runtime.h>
#include <hip/hip_bf16.h>
#include <hip/hip_fp16.h>

#define N_NODES 10000
#define N_EDGES 320000
#define IN_DIM  256
#define HID     64
#define HEADS   4
#define D1      (HEADS * HID)   // 256
#define OUT_DIM 64
#define NEG_SLOPE 0.2f

typedef __attribute__((ext_vector_type(16))) _Float16 v16h;
typedef __attribute__((ext_vector_type(8)))  _Float16 v8h;
typedef __attribute__((ext_vector_type(8)))  float    v8f;

// ---------------------------------------------------------------------------
// float atomic max via signed/unsigned bit-order trick (native 32-bit atomics)
// ---------------------------------------------------------------------------
__device__ inline void atomicMaxF(float* addr, float val) {
    if (val >= 0.0f)
        atomicMax((int*)addr, __float_as_int(val));
    else
        atomicMin((unsigned int*)addr, (unsigned int)__float_as_int(val));
}

// ---------------------------------------------------------------------------
// elementwise helpers
// ---------------------------------------------------------------------------
__global__ void cvt_f32_to_f16(const float* __restrict__ in, _Float16* __restrict__ out, int n) {
    int i = blockIdx.x * blockDim.x + threadIdx.x;
    if (i < n) out[i] = (_Float16)in[i];
}

// W[K,N] (f32, row-major) -> Wt[N,K] (f16, row-major)
__global__ void cvt_transpose_f16(const float* __restrict__ W, _Float16* __restrict__ Wt,
                                  int K, int N) {
    int i = blockIdx.x * blockDim.x + threadIdx.x;
    if (i >= K * N) return;
    int k = i / N, n = i % N;
    Wt[(size_t)n * K + k] = (_Float16)W[i];
}

__global__ void fill_f32(float* __restrict__ p, float v, int n) {
    int i = blockIdx.x * blockDim.x + threadIdx.x;
    if (i < n) p[i] = v;
}

// out[n*d + j] = bias[j]
__global__ void init_bias(float* __restrict__ out, const float* __restrict__ bias, int n, int d) {
    int i = blockIdx.x * blockDim.x + threadIdx.x;
    if (i < n * d) out[i] = bias[i % d];
}

// h1 = ELU(out + bias) ; emit f16 copy for next-layer WMMA
__global__ void elu_bias_to_f16(const float* __restrict__ in, const float* __restrict__ bias,
                                _Float16* __restrict__ out, int n, int F) {
    int i = blockIdx.x * blockDim.x + threadIdx.x;
    if (i >= n) return;
    float x = in[i] + bias[i % F];
    x = (x > 0.0f) ? x : (__expf(x) - 1.0f);
    out[i] = (_Float16)x;
}

// ---------------------------------------------------------------------------
// WMMA fragment load: per ISA 16-bit A/B 16x32 layout, lane owns two 8-element
// runs of K: elements 0..7 = K + 8*half, elements 8..15 = K + 16 + 8*half.
// Two aligned 16-byte loads -> global_load_b128.
// ---------------------------------------------------------------------------
__device__ inline v16h load_frag(const _Float16* __restrict__ base, int half) {
    v8h lo = *(const v8h*)(base + 8 * half);
    v8h hi = *(const v8h*)(base + 16 + 8 * half);
    v16h r;
#pragma unroll
    for (int i = 0; i < 8; ++i) { r[i] = lo[i]; r[8 + i] = hi[i]; }
    return r;
}

// ---------------------------------------------------------------------------
// WMMA GEMM: C[M,N] = A[M,K] * Bt[N,K]^T, f16 in / f32 out.
// Each wave computes a 16x64 strip (4 accumulators); one A fragment feeds
// 4 WMMAs per k-step. n0 = (blockIdx.y * wavesPerBlock + wave) * 64.
// Requires M%16==0, K%32==0, N%64==0 over the launched range.
// ---------------------------------------------------------------------------
__global__ void gemm_wmma_f16t(const _Float16* __restrict__ A, const _Float16* __restrict__ Bt,
                               float* __restrict__ C, int M, int N, int K) {
    const int lane = threadIdx.x & 31;
    const int wave = threadIdx.x >> 5;
    const int m0   = blockIdx.x * 16;
    const int n0   = (blockIdx.y * (blockDim.x >> 5) + wave) * 64;
    if (m0 >= M || n0 >= N) return;

    const int half = lane >> 4;           // 0: lanes 0-15, 1: lanes 16-31
    const int l15  = lane & 15;

    const _Float16* aptr = A + (size_t)(m0 + l15) * K;
    const _Float16* bptr0 = Bt + (size_t)(n0 +  0 + l15) * K;
    const _Float16* bptr1 = Bt + (size_t)(n0 + 16 + l15) * K;
    const _Float16* bptr2 = Bt + (size_t)(n0 + 32 + l15) * K;
    const _Float16* bptr3 = Bt + (size_t)(n0 + 48 + l15) * K;

    v8f acc0 = {}, acc1 = {}, acc2 = {}, acc3 = {};

    for (int k0 = 0; k0 < K; k0 += 32) {
        v16h a  = load_frag(aptr  + k0, half);
        v16h b0 = load_frag(bptr0 + k0, half);
        v16h b1 = load_frag(bptr1 + k0, half);
        v16h b2 = load_frag(bptr2 + k0, half);
        v16h b3 = load_frag(bptr3 + k0, half);
        acc0 = __builtin_amdgcn_wmma_f32_16x16x32_f16(false, a, false, b0, (short)0, acc0, false, false);
        acc1 = __builtin_amdgcn_wmma_f32_16x16x32_f16(false, a, false, b1, (short)0, acc1, false, false);
        acc2 = __builtin_amdgcn_wmma_f32_16x16x32_f16(false, a, false, b2, (short)0, acc2, false, false);
        acc3 = __builtin_amdgcn_wmma_f32_16x16x32_f16(false, a, false, b3, (short)0, acc3, false, false);
    }

    // D layout: VGPR r -> row m0 + r + 8*half, col = tile_n0 + (lane & 15)
    float* cbase = C + (size_t)(m0 + 8 * half) * N + n0 + l15;
#pragma unroll
    for (int r = 0; r < 8; ++r) {
        float* row = cbase + (size_t)r * N;
        row[0]  = acc0[r];
        row[16] = acc1[r];
        row[32] = acc2[r];
        row[48] = acc3[r];
    }
}

// ---------------------------------------------------------------------------
// el[n,h] = dot(feat[n,h,:], al[h,:]);  er likewise  (thread per (node,head))
// ---------------------------------------------------------------------------
__global__ void attn_scores(const float* __restrict__ feat, const float* __restrict__ al,
                            const float* __restrict__ ar, float* __restrict__ el,
                            float* __restrict__ er, int nheads, int d) {
    int idx = blockIdx.x * blockDim.x + threadIdx.x;
    if (idx >= N_NODES * nheads) return;
    int n = idx / nheads, h = idx % nheads;
    const float* f = feat + (size_t)(n * nheads + h) * d;
    float sl = 0.f, sr = 0.f;
    for (int j = 0; j < d; ++j) {
        float fv = f[j];
        sl += fv * al[h * d + j];
        sr += fv * ar[h * d + j];
    }
    el[idx] = sl;
    er[idx] = sr;
}

// ---------------------------------------------------------------------------
// edge softmax pass 1: e = leaky_relu(el[src]+er[dst]); stash e; segment max
// ---------------------------------------------------------------------------
__global__ void edge_max(const int* __restrict__ src, const int* __restrict__ dst,
                         const float* __restrict__ el, const float* __restrict__ er,
                         float* __restrict__ ebuf, float* __restrict__ m, int nheads) {
    int idx = blockIdx.x * blockDim.x + threadIdx.x;
    if (idx >= N_EDGES * nheads) return;
    int e = idx / nheads, h = idx % nheads;
    int s = src[e], t = dst[e];
    float x = el[s * nheads + h] + er[t * nheads + h];
    x = (x > 0.0f) ? x : NEG_SLOPE * x;
    ebuf[idx] = x;
    atomicMaxF(&m[t * nheads + h], x);
}

// pass 2: ex = exp(e - m[dst]); stash ex; segment sum denom
__global__ void edge_exp(const int* __restrict__ dst, const float* __restrict__ m,
                         float* __restrict__ ebuf, float* __restrict__ denom, int nheads) {
    int idx = blockIdx.x * blockDim.x + threadIdx.x;
    if (idx >= N_EDGES * nheads) return;
    int e = idx / nheads, h = idx % nheads;
    int t = dst[e];
    float ex = __expf(ebuf[idx] - m[t * nheads + h]);
    ebuf[idx] = ex;
    atomicAdd(&denom[t * nheads + h], ex);
}

// pass 3: out[dst] += feat[src] * alpha   (one wave32 per edge, coalesced)
__global__ void edge_aggregate(const int* __restrict__ src, const int* __restrict__ dst,
                               const float* __restrict__ feat, const float* __restrict__ ebuf,
                               const float* __restrict__ denom, float* __restrict__ out,
                               int nheads, int d) {
    int gwave = (blockIdx.x * blockDim.x + threadIdx.x) >> 5;
    int lane  = threadIdx.x & 31;
    if (gwave >= N_EDGES) return;
    int s = src[gwave], t = dst[gwave];
    int F = nheads * d;
    for (int h = 0; h < nheads; ++h) {
        float alpha = ebuf[gwave * nheads + h] / denom[t * nheads + h];
        for (int j = lane; j < d; j += 32) {
            atomicAdd(&out[(size_t)t * F + h * d + j],
                      feat[(size_t)s * F + h * d + j] * alpha);
        }
    }
}

// ---------------------------------------------------------------------------
// launch
// ---------------------------------------------------------------------------
extern "C" void kernel_launch(void* const* d_in, const int* in_sizes, int n_in,
                              void* d_out, int out_size, void* d_ws, size_t ws_size,
                              hipStream_t stream) {
    const float* h   = (const float*)d_in[0];
    const int*   src = (const int*)  d_in[1];
    const int*   dst = (const int*)  d_in[2];
    const float* W1  = (const float*)d_in[3];
    const float* al1 = (const float*)d_in[4];
    const float* ar1 = (const float*)d_in[5];
    const float* b1  = (const float*)d_in[6];
    const float* W2  = (const float*)d_in[7];
    const float* al2 = (const float*)d_in[8];
    const float* ar2 = (const float*)d_in[9];
    const float* b2  = (const float*)d_in[10];
    float* out = (float*)d_out;

    // carve workspace
    size_t off = 0;
    auto alloc = [&](size_t bytes) {
        void* p = (char*)d_ws + off;
        off += (bytes + 255) & ~(size_t)255;
        return p;
    };
    _Float16* hf16   = (_Float16*)alloc((size_t)N_NODES * IN_DIM  * 2);
    _Float16* W1t16  = (_Float16*)alloc((size_t)IN_DIM  * D1      * 2);  // [D1, IN_DIM]
    _Float16* W2t16  = (_Float16*)alloc((size_t)D1      * OUT_DIM * 2);  // [OUT_DIM, D1]
    float*    feat1  = (float*)   alloc((size_t)N_NODES * D1      * 4);
    float*    el1    = (float*)   alloc((size_t)N_NODES * HEADS   * 4);
    float*    er1    = (float*)   alloc((size_t)N_NODES * HEADS   * 4);
    float*    m1     = (float*)   alloc((size_t)N_NODES * HEADS   * 4);
    float*    den1   = (float*)   alloc((size_t)N_NODES * HEADS   * 4);
    float*    ebuf1  = (float*)   alloc((size_t)N_EDGES * HEADS   * 4);
    float*    out1   = (float*)   alloc((size_t)N_NODES * D1      * 4);
    _Float16* h1f16  = (_Float16*)alloc((size_t)N_NODES * D1      * 2);
    float*    feat2  = (float*)   alloc((size_t)N_NODES * OUT_DIM * 4);
    float*    el2    = (float*)   alloc((size_t)N_NODES * 4);
    float*    er2    = (float*)   alloc((size_t)N_NODES * 4);
    float*    m2     = (float*)   alloc((size_t)N_NODES * 4);
    float*    den2   = (float*)   alloc((size_t)N_NODES * 4);
    float*    ebuf2  = (float*)   alloc((size_t)N_EDGES * 4);

    const float NEG_INF = -__builtin_inff();
    auto cdiv = [](int a, int b) { return (a + b - 1) / b; };

    // ---- precision conversion (+ weight transpose for b128 fragment loads) ----
    cvt_f32_to_f16<<<cdiv(N_NODES * IN_DIM, 256), 256, 0, stream>>>(h, hf16, N_NODES * IN_DIM);
    cvt_transpose_f16<<<cdiv(IN_DIM * D1, 256), 256, 0, stream>>>(W1, W1t16, IN_DIM, D1);
    cvt_transpose_f16<<<cdiv(D1 * OUT_DIM, 256), 256, 0, stream>>>(W2, W2t16, D1, OUT_DIM);

    // ================= layer 1 =================
    // feat1 = h @ W1   (10000x256 @ 256x256): 4 waves/block, 16x64 strip per wave
    gemm_wmma_f16t<<<dim3(N_NODES / 16, 1), 128, 0, stream>>>(hf16, W1t16, feat1,
                                                              N_NODES, D1, IN_DIM);
    attn_scores<<<cdiv(N_NODES * HEADS, 256), 256, 0, stream>>>(feat1, al1, ar1, el1, er1,
                                                                HEADS, HID);
    fill_f32<<<cdiv(N_NODES * HEADS, 256), 256, 0, stream>>>(m1,   NEG_INF, N_NODES * HEADS);
    fill_f32<<<cdiv(N_NODES * HEADS, 256), 256, 0, stream>>>(den1, 0.0f,    N_NODES * HEADS);
    fill_f32<<<cdiv(N_NODES * D1,    256), 256, 0, stream>>>(out1, 0.0f,    N_NODES * D1);

    edge_max<<<cdiv(N_EDGES * HEADS, 256), 256, 0, stream>>>(src, dst, el1, er1, ebuf1, m1, HEADS);
    edge_exp<<<cdiv(N_EDGES * HEADS, 256), 256, 0, stream>>>(dst, m1, ebuf1, den1, HEADS);
    edge_aggregate<<<cdiv(N_EDGES * 32, 256), 256, 0, stream>>>(src, dst, feat1, ebuf1, den1,
                                                                out1, HEADS, HID);
    // h1 = ELU(out1 + b1), as f16 for next GEMM
    elu_bias_to_f16<<<cdiv(N_NODES * D1, 256), 256, 0, stream>>>(out1, b1, h1f16,
                                                                 N_NODES * D1, D1);

    // ================= layer 2 =================
    // feat2 = h1 @ W2   (10000x256 @ 256x64): one wave covers the 64-col strip
    gemm_wmma_f16t<<<dim3(N_NODES / 16, 1), 32, 0, stream>>>(h1f16, W2t16, feat2,
                                                             N_NODES, OUT_DIM, D1);
    attn_scores<<<cdiv(N_NODES, 256), 256, 0, stream>>>(feat2, al2, ar2, el2, er2, 1, OUT_DIM);
    fill_f32<<<cdiv(N_NODES, 256), 256, 0, stream>>>(m2,   NEG_INF, N_NODES);
    fill_f32<<<cdiv(N_NODES, 256), 256, 0, stream>>>(den2, 0.0f,    N_NODES);
    init_bias<<<cdiv(N_NODES * OUT_DIM, 256), 256, 0, stream>>>(out, b2, N_NODES, OUT_DIM);

    edge_max<<<cdiv(N_EDGES, 256), 256, 0, stream>>>(src, dst, el2, er2, ebuf2, m2, 1);
    edge_exp<<<cdiv(N_EDGES, 256), 256, 0, stream>>>(dst, m2, ebuf2, den2, 1);
    edge_aggregate<<<cdiv(N_EDGES * 32, 256), 256, 0, stream>>>(src, dst, feat2, ebuf2, den2,
                                                                out, 1, OUT_DIM);
    (void)in_sizes; (void)n_in; (void)out_size; (void)ws_size;
}
